// TemplateFeatAligner_82617990906034
// MI455X (gfx1250) — matrix operand order repaired and checked by reading, not compiled, not run on hardware
//
#include <hip/hip_runtime.h>
#include <hip/hip_bf16.h>
#include <stdint.h>

// Problem constants (match reference)
#define Bn   16
#define Qn   500
#define Tn   24
#define Cn   256
#define Hn   64
#define Wn   64
#define NHn  8
#define DHn  32
#define QG   2          // queries per fused block (same b, consecutive q)
#define KV_STRIDE 260   // 256 + 4 pad floats: breaks LDS bank conflicts on row-strided reads

typedef __attribute__((ext_vector_type(16))) __bf16 v16bf;
typedef __attribute__((ext_vector_type(8)))  float  v8f;

__device__ __forceinline__ unsigned short f2bf(float f) {
  union { float f; unsigned int u; } x; x.f = f;
  unsigned int u = x.u;
  u += 0x7fffu + ((u >> 16) & 1u);   // round-to-nearest-even
  return (unsigned short)(u >> 16);
}

// Async copy of 16 bytes/lane from global (GVS: SGPR base + 32-bit VGPR byte
// offset) into LDS (per-lane LDS byte address in VGPR).  Tracked by ASYNCcnt.
__device__ __forceinline__ void async_b128(void* lds_ptr, const void* gbase, int byte_off) {
  unsigned lds_off = (unsigned)(uintptr_t)lds_ptr;   // low 32 bits = LDS offset
  asm volatile("global_load_async_to_lds_b128 %0, %1, %2"
               :: "v"(lds_off), "v"(byte_off), "s"(gbase) : "memory");
}
__device__ __forceinline__ void wait_async0() {
  asm volatile("s_wait_asynccnt 0x0" ::: "memory");
}

// ---------------------------------------------------------------------------
// Kernel 1: BCHW -> BHWC transpose (32x32 LDS tiles, coalesced both ways).
// ---------------------------------------------------------------------------
__global__ __launch_bounds__(256) void transpose_bchw_bhwc_kernel(
    const float* __restrict__ in, float* __restrict__ out) {
  __shared__ float tile[32][33];
  int b  = blockIdx.z;
  int ct = blockIdx.y;   // channel tile (256/32 = 8)
  int pt = blockIdx.x;   // pixel tile   (4096/32 = 128)
  int tx = threadIdx.x;  // 0..31
  int ty = threadIdx.y;  // 0..7
  const float* src = in  + (size_t)b * Cn * Hn * Wn;
  float*       dst = out + (size_t)b * Hn * Wn * Cn;
#pragma unroll
  for (int i = 0; i < 32; i += 8) {
    int cc = ct * 32 + ty + i;
    int p  = pt * 32 + tx;
    tile[ty + i][tx] = src[(size_t)cc * (Hn * Wn) + p];
  }
  __syncthreads();
#pragma unroll
  for (int i = 0; i < 32; i += 8) {
    int p  = pt * 32 + ty + i;
    int cc = ct * 32 + tx;
    dst[(size_t)p * Cn + cc] = tile[tx][ty + i];
  }
}

// ---------------------------------------------------------------------------
// Kernel 2: pack Wq/Wk/Wv into WMMA B-operand tile layout (bf16).
// B 32x16 tile: lane group g, element e holds K = g*8 + e (e<8),
// K = 16 + g*8 + (e-8) (e>=8).  Flat: [ntile][kstep][lane][e].
// ---------------------------------------------------------------------------
__global__ __launch_bounds__(256) void pack_weights_kernel(
    const float* __restrict__ Wq, const float* __restrict__ Wk,
    const float* __restrict__ Wv,
    unsigned short* __restrict__ WqT, unsigned short* __restrict__ WkT,
    unsigned short* __restrict__ WvT) {
  int idx = blockIdx.x * 256 + threadIdx.x;   // 0 .. 3*65536
  int w = idx >> 16;
  int s = idx & 0xFFFF;
  int e    = s & 15;
  int lane = (s >> 4) & 31;
  int ks   = (s >> 9) & 7;
  int nt   = s >> 12;
  int g  = lane >> 4;
  int nl = lane & 15;
  int kg = ks * 32 + ((e < 8) ? (g * 8 + e) : (16 + g * 8 + (e - 8)));
  int n  = nt * 16 + nl;
  const float* src = (w == 0) ? Wq : (w == 1) ? Wk : Wv;
  unsigned short* dst = (w == 0) ? WqT : (w == 1) ? WkT : WvT;
  dst[s] = f2bf(src[(size_t)kg * Cn + n]);
}

// ---------------------------------------------------------------------------
// Kernel 3: temp_proj = sigmoid(temp_feat @ Wt + bt)   (384x256, tiny)
// ---------------------------------------------------------------------------
__global__ __launch_bounds__(256) void temp_proj_kernel(
    const float* __restrict__ temp_feat, const float* __restrict__ Wt,
    const float* __restrict__ bt, float* __restrict__ tp) {
  __shared__ float row[Cn];
  int r = blockIdx.x;        // b*T + t
  int c = threadIdx.x;
  row[c] = temp_feat[(size_t)r * Cn + c];
  __syncthreads();
  float s = bt[c];
  for (int k = 0; k < Cn; ++k) s += row[k] * Wt[(size_t)k * Cn + c];
  tp[(size_t)r * Cn + c] = 1.0f / (1.0f + __expf(-s));
}

// ---------------------------------------------------------------------------
// Kernel 4: roi_align 1x1 (aligned=True, sampling_ratio=-1, adaptive grid).
// ---------------------------------------------------------------------------
__global__ __launch_bounds__(256) void roi_align_kernel(
    const float* __restrict__ featT, const float* __restrict__ ref_points,
    const float* __restrict__ image_sizes, float* __restrict__ roi) {
  int blk = blockIdx.x;            // b*Q + q
  int b   = blk / Qn;
  int c   = threadIdx.x;
  const float* rp = ref_points + (size_t)blk * 4;
  float hs  = image_sizes[b * 2 + 0];
  float wsz = image_sizes[b * 2 + 1];
  float cx = rp[0] * hs, cy = rp[1] * wsz, bw = rp[2] * hs, bh = rp[3] * wsz;
  float x1 = (cx - 0.5f * bw) - 0.5f;
  float y1 = (cy - 0.5f * bh) - 0.5f;
  float roiw = bw, roih = bh;
  float gh = fminf(fmaxf(ceilf(roih), 1.0f), 8.0f);
  float gw = fminf(fmaxf(ceilf(roiw), 1.0f), 8.0f);
  int ghi = (int)gh, gwi = (int)gw;
  const float* fb = featT + (size_t)b * Hn * Wn * Cn;
  float acc = 0.0f;
  for (int iy = 0; iy < ghi; ++iy) {
    float yy = y1 + ((float)iy + 0.5f) * roih / gh;
    for (int ix = 0; ix < gwi; ++ix) {
      float xx = x1 + ((float)ix + 0.5f) * roiw / gw;
      bool valid = (yy > -1.0f) && (yy < (float)Hn) && (xx > -1.0f) && (xx < (float)Wn);
      float y = fminf(fmaxf(yy, 0.0f), (float)(Hn - 1));
      float x = fminf(fmaxf(xx, 0.0f), (float)(Wn - 1));
      int y0 = min((int)floorf(y), Hn - 1);
      int x0 = min((int)floorf(x), Wn - 1);
      int y1i = min(y0 + 1, Hn - 1);
      int x1i = min(x0 + 1, Wn - 1);
      float ly = y - (float)y0, lx = x - (float)x0;
      float v00 = fb[((size_t)(y0 * Wn) + x0) * Cn + c];
      float v01 = fb[((size_t)(y0 * Wn) + x1i) * Cn + c];
      float v10 = fb[((size_t)(y1i * Wn) + x0) * Cn + c];
      float v11 = fb[((size_t)(y1i * Wn) + x1i) * Cn + c];
      float val = (1.0f - ly) * (1.0f - lx) * v00 + (1.0f - ly) * lx * v01 +
                  ly * (1.0f - lx) * v10 + ly * lx * v11;
      acc += valid ? val : 0.0f;
    }
  }
  roi[(size_t)blk * Cn + c] = acc / (gh * gw);
}

// A-fragment mapping (ISA 7.12.2, 16-bit A 16x32): lane 0-15 = row M, lane
// group g selects K offset {0,8}; K = (v>=4?16:0) + g*8 + (v&3)*2 + h.

// ---------------------------------------------------------------------------
// Kernel 5: q = roi_feat @ Wq + bq  via WMMA (16-row M tiles, N=256).
// ---------------------------------------------------------------------------
__global__ __launch_bounds__(128) void qproj_kernel(
    const float* __restrict__ rf, const unsigned short* __restrict__ WqT,
    const float* __restrict__ bq, float* __restrict__ qout) {
  __shared__ __align__(32) unsigned short A_us[8 * 32 * 16];   // 8 KB
  int rowbase = blockIdx.x * 16;
  int tid = threadIdx.x, lane = tid & 31, wv = tid >> 5;
  for (int s = tid; s < 4096; s += 128) {
    int e = s & 15, ln = (s >> 4) & 31, ks = (s >> 9) & 7;
    int g = ln >> 4, nl = ln & 15;
    int v = e >> 1, hb = e & 1;
    int kin = ((v & 4) ? 16 : 0) + g * 8 + (v & 3) * 2 + hb;
    int kg = ks * 32 + kin;
    A_us[s] = f2bf(rf[(size_t)(rowbase + nl) * Cn + kg]);
  }
  __syncthreads();
  int ntb = wv * 4;
  v8f zero = {0.f, 0.f, 0.f, 0.f, 0.f, 0.f, 0.f, 0.f};
  v8f acc[4] = {zero, zero, zero, zero};
#pragma unroll
  for (int ks = 0; ks < 8; ++ks) {
    v16bf a = *(const v16bf*)&A_us[ks * 512 + lane * 16];
#pragma unroll
    for (int j = 0; j < 4; ++j) {
      v16bf bfr = *(const v16bf*)&WqT[((size_t)(ntb + j) * 8 + ks) * 512 + lane * 16];
      acc[j] = __builtin_amdgcn_wmma_f32_16x16x32_bf16(false, a, false, bfr,
                                                       (short)0, acc[j], false, false);
    }
  }
  int nl = lane & 15, g = lane >> 4;
#pragma unroll
  for (int j = 0; j < 4; ++j) {
    int col = (ntb + j) * 16 + nl;
    float bb = bq[col];
#pragma unroll
    for (int r = 0; r < 8; ++r) {
      int m = g * 8 + r;   // C/D layout: lanes 0-15 -> M=r, lanes 16-31 -> M=r+8
      qout[(size_t)(rowbase + m) * Cn + col] = acc[j][r] + bb;
    }
  }
}

// ---------------------------------------------------------------------------
// Dual-query GEMM: two 32x256 bf16 A matrices (LDS tile layout) x packed
// 256x256 weight + bias -> two KV matrices (f32 LDS, rows < T).
// Each B fragment load now feeds 4 WMMAs (2 queries x 2 m-tiles).
// Per wave: 8 ksteps x 4 ntiles x 4 = 128 v_wmma.
// ---------------------------------------------------------------------------
__device__ __forceinline__ void gemm2_a_w_to_kv(
    const unsigned short* A0, const unsigned short* A1,
    const unsigned short* WT, const float* bias,
    float* KV0, float* KV1, int lane, int wv) {
  int ntb = wv * 4;
  v8f zero = {0.f, 0.f, 0.f, 0.f, 0.f, 0.f, 0.f, 0.f};
  v8f acc[16] = {zero, zero, zero, zero, zero, zero, zero, zero,
                 zero, zero, zero, zero, zero, zero, zero, zero};
#pragma unroll
  for (int ks = 0; ks < 8; ++ks) {
    v16bf a00 = *(const v16bf*)&A0[ks * 512 + lane * 16];         // q0, m-tile 0
    v16bf a01 = *(const v16bf*)&A0[(8 + ks) * 512 + lane * 16];   // q0, m-tile 1
    v16bf a10 = *(const v16bf*)&A1[ks * 512 + lane * 16];         // q1, m-tile 0
    v16bf a11 = *(const v16bf*)&A1[(8 + ks) * 512 + lane * 16];   // q1, m-tile 1
#pragma unroll
    for (int j = 0; j < 4; ++j) {
      v16bf bfr = *(const v16bf*)&WT[((size_t)(ntb + j) * 8 + ks) * 512 + lane * 16];
      acc[4 * j + 0] = __builtin_amdgcn_wmma_f32_16x16x32_bf16(false, a00, false, bfr,
                                                               (short)0, acc[4 * j + 0], false, false);
      acc[4 * j + 1] = __builtin_amdgcn_wmma_f32_16x16x32_bf16(false, a01, false, bfr,
                                                               (short)0, acc[4 * j + 1], false, false);
      acc[4 * j + 2] = __builtin_amdgcn_wmma_f32_16x16x32_bf16(false, a10, false, bfr,
                                                               (short)0, acc[4 * j + 2], false, false);
      acc[4 * j + 3] = __builtin_amdgcn_wmma_f32_16x16x32_bf16(false, a11, false, bfr,
                                                               (short)0, acc[4 * j + 3], false, false);
    }
  }
  int nl = lane & 15, g = lane >> 4;
#pragma unroll
  for (int j = 0; j < 4; ++j) {
    int col = (ntb + j) * 16 + nl;
    float bb = bias[col];
#pragma unroll
    for (int r = 0; r < 8; ++r) {
      int m0 = g * 8 + r;                        // rows 0..15: always < 24
      KV0[m0 * KV_STRIDE + col] = acc[4 * j + 0][r] + bb;
      KV1[m0 * KV_STRIDE + col] = acc[4 * j + 2][r] + bb;
      int m1 = 16 + m0;                          // rows 16..31: keep < 24
      if (m1 < Tn) {
        KV0[m1 * KV_STRIDE + col] = acc[4 * j + 1][r] + bb;
        KV1[m1 * KV_STRIDE + col] = acc[4 * j + 3][r] + bb;
      }
    }
  }
}

// ---------------------------------------------------------------------------
// Kernel 6: fused per-(b, q-pair) attention.
//   async-stage tp[b] slice (24 KB) + rf/q rows (2x2 KB) into LDS,
//   A_i = tp[b] (x) rf[b,qi]  ->  K = A@Wk+bk (WMMA) -> logits -> softmax
//   -> V = A@Wv+bv (WMMA, reusing K LDS) -> out = attn @ V.
// ---------------------------------------------------------------------------
__global__ __launch_bounds__(128) void fused_attn_kernel(
    const float* __restrict__ tp, const float* __restrict__ rf,
    const float* __restrict__ qproj,
    const unsigned short* __restrict__ WkT, const unsigned short* __restrict__ WvT,
    const float* __restrict__ bk, const float* __restrict__ bv,
    float* __restrict__ out) {
  __shared__ __align__(32) unsigned short A_us[QG][8192];   // 32 KB
  __shared__ float KV[QG][Tn * KV_STRIDE];                  // ~48.8 KB
  __shared__ __align__(16) float tp_s[Tn * Cn];             // 24 KB
  __shared__ __align__(16) float rf_s[QG * Cn];             // 2 KB (rows contiguous)
  __shared__ __align__(16) float qs[QG * Cn];               // 2 KB
  __shared__ float att[QG][NHn * Tn];

  int blk = blockIdx.x;               // 0 .. B*Q/QG-1
  int b   = blk / (Qn / QG);
  int qp  = blk % (Qn / QG);
  size_t row0 = (size_t)b * Qn + (size_t)qp * QG;   // first query row; QG rows contiguous
  int tid = threadIdx.x, lane = tid & 31, wv = tid >> 5;

  // ---- async staging (ASYNCcnt): tp slice + rf rows + q rows ----
  const float* tpg = tp + (size_t)b * Tn * Cn;
#pragma unroll
  for (int i = 0; i < (Tn * Cn) / (128 * 4); ++i) {      // 12 issues of 16B/lane
    int chunk = tid + i * 128;
    async_b128(&tp_s[chunk * 4], tpg, chunk * 16);
  }
  async_b128(&rf_s[tid * 4], rf + row0 * Cn, tid * 16);  // QG*256 f32 = 2 KB
  async_b128(&qs[tid * 4], qproj + row0 * Cn, tid * 16);
  wait_async0();
  __syncthreads();

  // ---- build both A matrices in bf16 WMMA A-fragment layout ----
  for (int s = tid; s < QG * 8192; s += 128) {
    int qg = s >> 13;
    int s2 = s & 8191;
    int e = s2 & 15, ln = (s2 >> 4) & 31, ks = (s2 >> 9) & 7, mt = s2 >> 12;
    int g = ln >> 4, nl = ln & 15;
    int v = e >> 1, hb = e & 1;
    int kin = ((v & 4) ? 16 : 0) + g * 8 + (v & 3) * 2 + hb;
    int kg = ks * 32 + kin;
    int m = mt * 16 + nl;
    float val = (m < Tn) ? tp_s[m * Cn + kg] * rf_s[qg * Cn + kg] : 0.0f;
    A_us[qg][s2] = f2bf(val);
  }
  __syncthreads();

  gemm2_a_w_to_kv(A_us[0], A_us[1], WkT, bk, KV[0], KV[1], lane, wv);  // K
  __syncthreads();

  const float scl = 0.17677669529663687f;        // 1/sqrt(DH)
  for (int p = tid; p < QG * NHn * Tn; p += 128) {
    int qg = p / (NHn * Tn);
    int pp = p % (NHn * Tn);
    int h = pp / Tn, t = pp % Tn;
    float s = 0.0f;
#pragma unroll
    for (int d = 0; d < DHn; ++d)
      s += qs[qg * Cn + h * DHn + d] * KV[qg][t * KV_STRIDE + h * DHn + d];
    att[qg][pp] = s * scl;
  }
  __syncthreads();

  if (tid < QG * NHn) {                           // softmax over T per (query, head)
    int qg = tid >> 3, h = tid & 7;
    float* ar = &att[qg][h * Tn];
    float mx = -3.0e38f;
    for (int t = 0; t < Tn; ++t) mx = fmaxf(mx, ar[t]);
    float sm = 0.0f;
    for (int t = 0; t < Tn; ++t) { float ev = __expf(ar[t] - mx); ar[t] = ev; sm += ev; }
    float inv = 1.0f / sm;
    for (int t = 0; t < Tn; ++t) ar[t] *= inv;
  }
  __syncthreads();

  gemm2_a_w_to_kv(A_us[0], A_us[1], WvT, bv, KV[0], KV[1], lane, wv);  // V (reuses K LDS)
  __syncthreads();

  for (int c = tid; c < QG * Cn; c += 128) {
    int qg = c >> 8;
    int cc = c & 255;
    int h = cc >> 5;
    float s = 0.0f;
#pragma unroll
    for (int t = 0; t < Tn; ++t) s += att[qg][h * Tn + t] * KV[qg][t * KV_STRIDE + cc];
    out[(row0 + qg) * Cn + cc] = s;
  }
}

// ---------------------------------------------------------------------------
// Workspace layout (bytes, all 32B-aligned):
//   featT  :        0 + 67108864   (B*H*W*C f32)
//   tp     : 67108864 + 393216     (B*T*C f32)
//   roi    : 67502080 + 8192000    (B*Q*C f32)
//   qproj  : 75694080 + 8192000    (B*Q*C f32)
//   WqT    : 83886080 + 131072     (bf16 tiles)
//   WkT    : 84017152 + 131072
//   WvT    : 84148224 + 131072     total ~84.3 MB
// ---------------------------------------------------------------------------
extern "C" void kernel_launch(void* const* d_in, const int* in_sizes, int n_in,
                              void* d_out, int out_size, void* d_ws, size_t ws_size,
                              hipStream_t stream) {
  (void)in_sizes; (void)n_in; (void)out_size; (void)ws_size;
  const float* image_feat  = (const float*)d_in[0];
  const float* temp_feat   = (const float*)d_in[1];
  const float* ref_points  = (const float*)d_in[2];
  const float* image_sizes = (const float*)d_in[3];
  const float* Wt = (const float*)d_in[4];
  const float* bt = (const float*)d_in[5];
  const float* Wq = (const float*)d_in[6];
  const float* bq = (const float*)d_in[7];
  const float* Wk = (const float*)d_in[8];
  const float* bk = (const float*)d_in[9];
  const float* Wv = (const float*)d_in[10];
  const float* bv = (const float*)d_in[11];
  float* out = (float*)d_out;

  char* ws = (char*)d_ws;
  float* featT = (float*)(ws + 0);
  float* tp    = (float*)(ws + 67108864);
  float* roi   = (float*)(ws + 67502080);
  float* qpr   = (float*)(ws + 75694080);
  unsigned short* WqT = (unsigned short*)(ws + 83886080);
  unsigned short* WkT = (unsigned short*)(ws + 84017152);
  unsigned short* WvT = (unsigned short*)(ws + 84148224);

  transpose_bchw_bhwc_kernel<<<dim3(128, 8, Bn), dim3(32, 8), 0, stream>>>(image_feat, featT);
  pack_weights_kernel<<<768, 256, 0, stream>>>(Wq, Wk, Wv, WqT, WkT, WvT);
  temp_proj_kernel<<<Bn * Tn, 256, 0, stream>>>(temp_feat, Wt, bt, tp);
  roi_align_kernel<<<Bn * Qn, 256, 0, stream>>>(featT, ref_points, image_sizes, roi);
  qproj_kernel<<<(Bn * Qn) / 16, 128, 0, stream>>>(roi, WqT, bq, qpr);
  fused_attn_kernel<<<(Bn * Qn) / QG, 128, 0, stream>>>(tp, roi, qpr, WkT, WvT, bk, bv, out);
}
